// TransformerStage_75376676044870
// MI455X (gfx1250) — compile-verified, matrix operand order checked by reading, not run on hardware
//
#include <hip/hip_runtime.h>
#include <math.h>

// ---------------- problem constants ----------------
#define B_    8
#define C_    384
#define H_    32
#define W_    32
#define HW_   1024
#define HEADS_ 12
#define G_    6
#define HC_   32
#define CG_   64
#define NGH_  2
#define D2_   1536
#define HK_   16
#define WK_   16
#define NS_   256
#define RPE_W 63
#define SCALE_ 0.17677669529663687f   // 32^-0.5

#define KSTRIDE 40   // bf16 row stride for K-major LDS tiles: 80B, 16B-multiple

typedef __attribute__((ext_vector_type(16))) __bf16 v16bf;
typedef __attribute__((ext_vector_type(8)))  float  v8f;
typedef __attribute__((ext_vector_type(4)))  float  f32x4;
typedef __attribute__((ext_vector_type(4)))  __bf16 bf16x4;

static __device__ __forceinline__ bf16x4 cvt4(f32x4 v) {
  bf16x4 r;
  r.x = (__bf16)v.x; r.y = (__bf16)v.y; r.z = (__bf16)v.z; r.w = (__bf16)v.w;
  return r;
}

static __device__ __forceinline__ float gelu_exact(float v) {
  return 0.5f * v * (1.0f + erff(v * 0.70710678118654752f));
}

static __device__ __forceinline__ v8f wmma_bf16(v16bf a, v16bf b, v8f c) {
  return __builtin_amdgcn_wmma_f32_16x16x32_bf16(
      false, a, false, b, (short)0, c, false, false);
}

// CDNA5 async global->LDS copies (per-lane), tracked by ASYNCcnt.
static __device__ __forceinline__ void async_copy_b32(unsigned lds_off,
                                                      const float* gptr) {
  asm volatile("global_load_async_to_lds_b32 %0, %1, off"
               :: "v"(lds_off), "v"(gptr) : "memory");
}
static __device__ __forceinline__ void async_copy_b128(unsigned lds_off,
                                                       const float* gptr) {
  asm volatile("global_load_async_to_lds_b128 %0, %1, off"
               :: "v"(lds_off), "v"(gptr) : "memory");
}
static __device__ __forceinline__ void wait_async0() {
  asm volatile("s_wait_asynccnt 0x0" ::: "memory");
}

// ---------------- depthwise 3x3 conv (+residual, optional exact-GELU) ------
__global__ void k_dwconv(const float* __restrict__ in, const float* __restrict__ wt,
                         const float* __restrict__ bs, float* __restrict__ out,
                         int Cch, int mode) {
  int idx = blockIdx.x * blockDim.x + threadIdx.x;
  int total = B_ * Cch * HW_;
  if (idx >= total) return;
  int w = idx & (W_ - 1);
  int h = (idx >> 5) & (H_ - 1);
  int c = (idx / HW_) % Cch;
  const float* plane = in + (size_t)(idx - (h * W_ + w));
  const float* ww = wt + c * 9;
  float acc = bs[c];
#pragma unroll
  for (int dy = 0; dy < 3; ++dy) {
    int iy = h + dy - 1;
    if (iy < 0 || iy >= H_) continue;
#pragma unroll
    for (int dx = 0; dx < 3; ++dx) {
      int ix = w + dx - 1;
      if (ix < 0 || ix >= W_) continue;
      acc += plane[iy * W_ + ix] * ww[dy * 3 + dx];
    }
  }
  float v = in[idx] + acc;
  out[idx] = mode ? gelu_exact(v) : v;
}

// ---------------- batched GEMM via WMMA bf16, software-pipelined ----------
// Y[b,m,n] = sum_k W[m,k] * X[b,k,n] + bias[m] (+ res[b,m,n])
// 256 thr = 8 waves (4 M x 2 N), tile 64x64, wave 16x32.
// Chunk k+1 is loaded into registers while WMMA consumes chunk k from LDS.
__global__ void __launch_bounds__(256)
k_gemm(const float* __restrict__ Wm, const float* __restrict__ X,
       const float* __restrict__ bias, const float* __restrict__ res,
       float* __restrict__ Y, int M, int N, int K) {
  __shared__ __bf16 sW[64 * 32];          // [m][k]
  __shared__ __bf16 sXT[64 * KSTRIDE];    // [n][k] (stride 40)
  int tid  = threadIdx.x;
  int lane = tid & 31;
  int wid  = tid >> 5;
  int wm = wid >> 1, wn = wid & 1;
  int bz = blockIdx.z;
  int m0 = blockIdx.y * 64;
  int n0 = blockIdx.x * 64;
  int hi = lane >> 4, l15 = lane & 15;

  const float* Wp = Wm + (size_t)m0 * K;          // 64 rows, K-contiguous
  const float* Xp = X + (size_t)bz * K * N + n0;  // 64 cols, n-contiguous rows

  // per-thread staging tasks (2 each): W quad + X k-quad
  int wt_m[2], wt_q[2], xt_n[2], xt_q[2];
#pragma unroll
  for (int i = 0; i < 2; ++i) {
    int idx = i * 256 + tid;
    wt_m[i] = idx >> 3;  wt_q[i] = idx & 7;   // 64 rows x 8 quads of k
    xt_n[i] = idx & 63;  xt_q[i] = idx >> 6;  // 64 cols x 8 quads of k
  }

  f32x4 wr[2]; float xr[2][4];
  // ---- preload chunk 0 into registers
#pragma unroll
  for (int i = 0; i < 2; ++i) {
    wr[i] = *(const f32x4*)(Wp + (size_t)wt_m[i] * K + wt_q[i] * 4);
    int base = (xt_q[i] * 4) * N + xt_n[i];
#pragma unroll
    for (int j = 0; j < 4; ++j) xr[i][j] = Xp[base + j * N];
  }

  v8f acc0 = {}; v8f acc1 = {};
  for (int kk = 0; kk < K; kk += 32) {
    // ---- stage registers -> LDS (packed bf16x4 = ds_store_b64)
#pragma unroll
    for (int i = 0; i < 2; ++i) {
      *(bf16x4*)&sW[wt_m[i] * 32 + wt_q[i] * 4] = cvt4(wr[i]);
      f32x4 xv; xv.x = xr[i][0]; xv.y = xr[i][1]; xv.z = xr[i][2]; xv.w = xr[i][3];
      *(bf16x4*)&sXT[xt_n[i] * KSTRIDE + xt_q[i] * 4] = cvt4(xv);
    }
    __syncthreads();
    // ---- issue next chunk's global loads (overlap with WMMA below)
    if (kk + 32 < K) {
#pragma unroll
      for (int i = 0; i < 2; ++i) {
        wr[i] = *(const f32x4*)(Wp + (size_t)wt_m[i] * K + kk + 32 + wt_q[i] * 4);
        int base = (kk + 32 + xt_q[i] * 4) * N + xt_n[i];
#pragma unroll
        for (int j = 0; j < 4; ++j) xr[i][j] = Xp[base + j * N];
      }
    }
    // ---- fragments (ds_load_b128 pairs) + WMMA
    v16bf a, b0, b1;
#pragma unroll
    for (int j = 0; j < 8; ++j) {
      int k0 = ((j >> 2) << 4) + (hi << 3) + ((j & 3) << 1);
      a[2 * j]     = sW[(wm * 16 + l15) * 32 + k0];
      a[2 * j + 1] = sW[(wm * 16 + l15) * 32 + k0 + 1];
      b0[2 * j]     = sXT[(wn * 32 + l15) * KSTRIDE + k0];
      b0[2 * j + 1] = sXT[(wn * 32 + l15) * KSTRIDE + k0 + 1];
      b1[2 * j]     = sXT[(wn * 32 + 16 + l15) * KSTRIDE + k0];
      b1[2 * j + 1] = sXT[(wn * 32 + 16 + l15) * KSTRIDE + k0 + 1];
    }
    acc0 = wmma_bf16(a, b0, acc0);
    acc1 = wmma_bf16(a, b1, acc1);
    __syncthreads();
  }
#pragma unroll
  for (int j = 0; j < 8; ++j) {
    int m = m0 + wm * 16 + j + (hi << 3);
    int n = n0 + wn * 32 + l15;
    size_t i0 = ((size_t)bz * M + m) * (size_t)N + n;
    float bsv = bias ? bias[m] : 0.f;
    float v0 = acc0[j] + bsv;
    float v1 = acc1[j] + bsv;
    if (res) { v0 += res[i0]; v1 += res[i0 + 16]; }
    Y[i0] = v0; Y[i0 + 16] = v1;
  }
}

// ---------------- offset network (dwconv s2 + LN + GELU + pw -> pos) -------
__global__ void k_offset(const float* __restrict__ Q, const float* __restrict__ odw,
                         const float* __restrict__ odb, const float* __restrict__ olw,
                         const float* __restrict__ olb, const float* __restrict__ opw,
                         float* __restrict__ POS) {
  int bg = blockIdx.x;        // 0..47
  int p  = threadIdx.x;       // 0..255 -> (hk,wk)
  int hk = p >> 4, wk = p & 15;
  float vals[CG_];
  float mean = 0.f;
  for (int c = 0; c < CG_; ++c) {
    float acc = odb[c];
    const float* plane = Q + ((size_t)bg * CG_ + c) * HW_;
    const float* ww = odw + c * 9;
#pragma unroll
    for (int dy = 0; dy < 3; ++dy) {
      int iy = 2 * hk + dy - 1;
      if (iy < 0 || iy >= H_) continue;
#pragma unroll
      for (int dx = 0; dx < 3; ++dx) {
        int ix = 2 * wk + dx - 1;
        if (ix < 0 || ix >= W_) continue;
        acc += plane[iy * W_ + ix] * ww[dy * 3 + dx];
      }
    }
    vals[c] = acc; mean += acc;
  }
  mean *= (1.f / CG_);
  float var = 0.f;
  for (int c = 0; c < CG_; ++c) { float d = vals[c] - mean; var += d * d; }
  var *= (1.f / CG_);
  float rstd = rsqrtf(var + 1e-5f);
  float oy = 0.f, ox = 0.f;
  for (int c = 0; c < CG_; ++c) {
    float g = gelu_exact((vals[c] - mean) * rstd * olw[c] + olb[c]);
    oy += opw[c] * g;
    ox += opw[CG_ + c] * g;
  }
  float ry = ((float)hk + 0.5f) * (2.f / 15.f) - 1.f;
  float rx = ((float)wk + 0.5f) * (2.f / 15.f) - 1.f;
  POS[((size_t)bg * NS_ + p) * 2 + 0] = fminf(fmaxf(oy + ry, -1.f), 1.f);
  POS[((size_t)bg * NS_ + p) * 2 + 1] = fminf(fmaxf(ox + rx, -1.f), 1.f);
}

// ---------------- deformable bilinear sampling -----------------------------
__global__ void k_sample(const float* __restrict__ X0, const float* __restrict__ POS,
                         float* __restrict__ XS) {
  int idx = blockIdx.x * blockDim.x + threadIdx.x;
  if (idx >= 48 * CG_ * NS_) return;
  int s  = idx & (NS_ - 1);
  int c  = (idx >> 8) & (CG_ - 1);
  int bg = idx >> 14;
  float py = POS[((size_t)bg * NS_ + s) * 2 + 0];
  float px = POS[((size_t)bg * NS_ + s) * 2 + 1];
  float gx = (px + 1.f) * 0.5f * (W_ - 1);
  float gy = (py + 1.f) * 0.5f * (H_ - 1);
  float x0f = floorf(gx), y0f = floorf(gy);
  float wx = gx - x0f, wy = gy - y0f;
  int x0 = min(max((int)x0f, 0), W_ - 1);
  int y0 = min(max((int)y0f, 0), H_ - 1);
  int x1 = min(x0 + 1, W_ - 1);
  int y1 = min(y0 + 1, H_ - 1);
  const float* pl = X0 + ((size_t)bg * CG_ + c) * HW_;
  float v00 = pl[y0 * W_ + x0], v01 = pl[y0 * W_ + x1];
  float v10 = pl[y1 * W_ + x0], v11 = pl[y1 * W_ + x1];
  XS[idx] = v00 * (1 - wx) * (1 - wy) + v01 * wx * (1 - wy)
          + v10 * (1 - wx) * wy + v11 * wx * wy;
}

// ---------------- fused flash-style deformable attention -------------------
// grid: (16 query tiles of 64, 96 batch*heads); block: 128 thr = 4 waves.
// V / POS / RPE staged by async-DMA (ASYNCcnt); K,Q transposed to K-major bf16
// via quad-vectorized DS staging. ~139KB dynamic LDS of 320KB WGP budget.
__global__ void __launch_bounds__(128)
k_attn(const float* __restrict__ Q, const float* __restrict__ Kb,
       const float* __restrict__ Vb, const float* __restrict__ POS,
       const float* __restrict__ rpe_d, float* __restrict__ AO) {
  extern __shared__ char smem[];
  float*  sS   = (float*)(smem);                 // 64x256 f32
  __bf16* sKT  = (__bf16*)(smem + 65536);        // [256][40] bf16
  __bf16* sQT  = (__bf16*)(smem + 86016);        // [64][40] bf16
  float*  sV   = (float*)(smem + 91136);         // [32][256] f32 (async)
  float*  sPos = (float*)(smem + 123904);        // [256][2] f32 (async)
  float*  sRow = (float*)(smem + 125952);        // [64] f32
  float*  sRpe = (float*)(smem + 126208);        // [63*63] f32 (async)

  int mt = blockIdx.x;
  int bh = blockIdx.y;
  int head = bh % HEADS_;
  int b = bh / HEADS_;
  int g = head / NGH_;
  int bg = b * G_ + g;
  const float* rpe_h = rpe_d + (size_t)head * RPE_W * RPE_W;

  int tid = threadIdx.x;
  int lane = tid & 31;
  int w = tid >> 5;
  int hi = lane >> 4, l15 = lane & 15;

  // ---- async DMA: V (b128), POS (b32), RPE (b32)
  {
    unsigned vbase = (unsigned)(size_t)(void*)sV;
    const float* vsrc = Vb + (size_t)bh * HC_ * NS_;
    for (int i = tid; i < HC_ * NS_ / 4; i += 128)
      async_copy_b128(vbase + (unsigned)i * 16u, vsrc + i * 4);
    unsigned pbase = (unsigned)(size_t)(void*)sPos;
    const float* psrc = POS + (size_t)bg * NS_ * 2;
    for (int i = tid; i < 2 * NS_; i += 128)
      async_copy_b32(pbase + (unsigned)i * 4u, psrc + i);
    unsigned rbase = (unsigned)(size_t)(void*)sRpe;
    for (int i = tid; i < RPE_W * RPE_W; i += 128)
      async_copy_b32(rbase + (unsigned)i * 4u, rpe_h + i);
  }
  // ---- transposing/converting staging (k-quads: 4 coalesced loads -> b64 store)
  {
    const float* ksrc = Kb + (size_t)bh * HC_ * NS_;
    for (int i = tid; i < (HC_ / 4) * NS_; i += 128) {
      int n = i & 255, k0 = (i >> 8) * 4;
      f32x4 v;
      v.x = ksrc[(k0 + 0) * NS_ + n];
      v.y = ksrc[(k0 + 1) * NS_ + n];
      v.z = ksrc[(k0 + 2) * NS_ + n];
      v.w = ksrc[(k0 + 3) * NS_ + n];
      *(bf16x4*)&sKT[n * KSTRIDE + k0] = cvt4(v);
    }
    const float* qsrc = Q + (size_t)bh * HC_ * HW_ + mt * 64;
    for (int i = tid; i < (HC_ / 4) * 64; i += 128) {
      int m = i & 63, k0 = (i >> 6) * 4;
      f32x4 v;
      v.x = qsrc[(k0 + 0) * HW_ + m];
      v.y = qsrc[(k0 + 1) * HW_ + m];
      v.z = qsrc[(k0 + 2) * HW_ + m];
      v.w = qsrc[(k0 + 3) * HW_ + m];
      *(bf16x4*)&sQT[m * KSTRIDE + k0] = cvt4(v);
    }
  }
  wait_async0();
  __syncthreads();

  // ---- scores = Q^T K * scale
  v16bf aq;
#pragma unroll
  for (int j = 0; j < 8; ++j) {
    int k0 = ((j >> 2) << 4) + (hi << 3) + ((j & 3) << 1);
    aq[2 * j]     = sQT[(w * 16 + l15) * KSTRIDE + k0];
    aq[2 * j + 1] = sQT[(w * 16 + l15) * KSTRIDE + k0 + 1];
  }
#pragma unroll
  for (int nt = 0; nt < 16; ++nt) {
    v16bf bk;
#pragma unroll
    for (int j = 0; j < 8; ++j) {
      int k0 = ((j >> 2) << 4) + (hi << 3) + ((j & 3) << 1);
      bk[2 * j]     = sKT[(nt * 16 + l15) * KSTRIDE + k0];
      bk[2 * j + 1] = sKT[(nt * 16 + l15) * KSTRIDE + k0 + 1];
    }
    v8f sc = {};
    sc = wmma_bf16(aq, bk, sc);
#pragma unroll
    for (int j = 0; j < 8; ++j) {
      int r = j + (hi << 3);
      sS[(w * 16 + r) * NS_ + nt * 16 + l15] = sc[j] * SCALE_;
    }
  }
  __syncthreads();

  // ---- RPE bias (LDS bilinear) + softmax; 2 lanes per row
  {
    int r = lane >> 1;
    int half = lane & 1;
    int mrow = w * 16 + r;
    int mg = mt * 64 + mrow;
    float qy = (float)(mg >> 5) * (2.f / 31.f) - 1.f;
    float qx = (float)(mg & 31) * (2.f / 31.f) - 1.f;
    float mx = -3.0e38f;
    for (int n = 0; n < 128; ++n) {
      int nn = half * 128 + n;
      float py = sPos[nn * 2], px = sPos[nn * 2 + 1];
      float dy = (qy - py) * 0.5f, dx = (qx - px) * 0.5f;
      float gx = (dx + 1.f) * 0.5f * (RPE_W - 1);
      float gy = (dy + 1.f) * 0.5f * (RPE_W - 1);
      float x0f = floorf(gx), y0f = floorf(gy);
      float wx = gx - x0f, wy = gy - y0f;
      int x0 = min(max((int)x0f, 0), RPE_W - 1);
      int y0 = min(max((int)y0f, 0), RPE_W - 1);
      int x1 = min(x0 + 1, RPE_W - 1);
      int y1 = min(y0 + 1, RPE_W - 1);
      float v00 = sRpe[y0 * RPE_W + x0], v01 = sRpe[y0 * RPE_W + x1];
      float v10 = sRpe[y1 * RPE_W + x0], v11 = sRpe[y1 * RPE_W + x1];
      float bias = v00 * (1 - wx) * (1 - wy) + v01 * wx * (1 - wy)
                 + v10 * (1 - wx) * wy + v11 * wx * wy;
      float s = sS[mrow * NS_ + nn] + bias;
      sS[mrow * NS_ + nn] = s;
      mx = fmaxf(mx, s);
    }
    mx = fmaxf(mx, __shfl_xor(mx, 1, 32));
    float sum = 0.f;
    for (int n = 0; n < 128; ++n) {
      int nn = half * 128 + n;
      float pv = __expf(sS[mrow * NS_ + nn] - mx);
      sS[mrow * NS_ + nn] = pv;
      sum += pv;
    }
    sum += __shfl_xor(sum, 1, 32);
    if (half == 0) sRow[mrow] = sum;
  }
  __syncthreads();

  // ---- out = P @ V^T  (8 K-chunks of 32 samples; 2 channel halves)
  v8f o0 = {}, o1 = {};
  for (int kc = 0; kc < 8; ++kc) {
    v16bf ap, bv0, bv1;
#pragma unroll
    for (int j = 0; j < 8; ++j) {
      int k0 = ((j >> 2) << 4) + (hi << 3) + ((j & 3) << 1);
      ap[2 * j]     = (__bf16)(sS[(w * 16 + l15) * NS_ + kc * 32 + k0]);
      ap[2 * j + 1] = (__bf16)(sS[(w * 16 + l15) * NS_ + kc * 32 + k0 + 1]);
      bv0[2 * j]     = (__bf16)(sV[l15 * NS_ + kc * 32 + k0]);
      bv0[2 * j + 1] = (__bf16)(sV[l15 * NS_ + kc * 32 + k0 + 1]);
      bv1[2 * j]     = (__bf16)(sV[(16 + l15) * NS_ + kc * 32 + k0]);
      bv1[2 * j + 1] = (__bf16)(sV[(16 + l15) * NS_ + kc * 32 + k0 + 1]);
    }
    o0 = wmma_bf16(ap, bv0, o0);
    o1 = wmma_bf16(ap, bv1, o1);
  }
#pragma unroll
  for (int j = 0; j < 8; ++j) {
    int r = j + (hi << 3);
    int mrow = w * 16 + r;
    float inv = 1.f / sRow[mrow];
    int mg = mt * 64 + mrow;
    AO[((size_t)bh * HC_ + l15) * HW_ + mg]      = o0[j] * inv;
    AO[((size_t)bh * HC_ + 16 + l15) * HW_ + mg] = o1[j] * inv;
  }
}

// ---------------- host orchestration ---------------------------------------
extern "C" void kernel_launch(void* const* d_in, const int* in_sizes, int n_in,
                              void* d_out, int out_size, void* d_ws, size_t ws_size,
                              hipStream_t stream) {
  (void)in_sizes; (void)n_in; (void)out_size; (void)ws_size;
  const float* x_in = (const float*)d_in[0];
  const float* lpu_w = (const float*)d_in[1];
  const float* lpu_b = (const float*)d_in[2];
  const float* wq = (const float*)d_in[3];  const float* bq = (const float*)d_in[4];
  const float* wk = (const float*)d_in[5];  const float* bk = (const float*)d_in[6];
  const float* wv = (const float*)d_in[7];  const float* bv = (const float*)d_in[8];
  const float* wo = (const float*)d_in[9];  const float* bo = (const float*)d_in[10];
  const float* odw = (const float*)d_in[11]; const float* odb = (const float*)d_in[12];
  const float* olw = (const float*)d_in[13]; const float* olb = (const float*)d_in[14];
  const float* opw = (const float*)d_in[15];
  const float* rpe = (const float*)d_in[16];
  const float* w1 = (const float*)d_in[17]; const float* b1 = (const float*)d_in[18];
  const float* mdw = (const float*)d_in[19]; const float* mdb = (const float*)d_in[20];
  const float* w2 = (const float*)d_in[21]; const float* b2 = (const float*)d_in[22];

  float* WS = (float*)d_ws;
  size_t o = 0;
  float* X0  = WS + o; o += (size_t)B_ * C_ * HW_;   // t0 = x + lpu(x)
  float* X1  = WS + o; o += (size_t)B_ * C_ * HW_;   // t1 = t0 + attn out
  float* X2  = WS + o; o += (size_t)B_ * C_ * HW_;   // depth output (non-final)
  float* Qb  = WS + o; o += (size_t)B_ * C_ * HW_;
  float* AOb = WS + o; o += (size_t)B_ * C_ * HW_;
  float* XSb = WS + o; o += (size_t)B_ * C_ * NS_;
  float* Kbf = WS + o; o += (size_t)B_ * C_ * NS_;
  float* Vbf = WS + o; o += (size_t)B_ * C_ * NS_;
  float* POSb= WS + o; o += (size_t)48 * NS_ * 2;
  float* Hb  = WS + o; o += (size_t)B_ * D2_ * HW_;
  float* H2b = WS + o; o += (size_t)B_ * D2_ * HW_;

  const size_t attn_smem = 142336;
  const float* xprev = x_in;

  for (int d = 0; d < 2; ++d) {
    int tot = B_ * C_ * HW_;
    k_dwconv<<<(tot + 255) / 256, 256, 0, stream>>>(
        xprev, lpu_w + (size_t)d * C_ * 9, lpu_b + (size_t)d * C_, X0, C_, 0);

    k_gemm<<<dim3(HW_ / 64, C_ / 64, B_), 256, 0, stream>>>(
        wq + (size_t)d * C_ * C_, X0, bq + d * C_, nullptr, Qb, C_, HW_, C_);

    k_offset<<<48, 256, 0, stream>>>(
        Qb, odw + (size_t)d * CG_ * 9, odb + d * CG_, olw + d * CG_,
        olb + d * CG_, opw + (size_t)d * 2 * CG_, POSb);

    int stot = 48 * CG_ * NS_;
    k_sample<<<(stot + 255) / 256, 256, 0, stream>>>(X0, POSb, XSb);

    k_gemm<<<dim3(NS_ / 64, C_ / 64, B_), 256, 0, stream>>>(
        wk + (size_t)d * C_ * C_, XSb, bk + d * C_, nullptr, Kbf, C_, NS_, C_);
    k_gemm<<<dim3(NS_ / 64, C_ / 64, B_), 256, 0, stream>>>(
        wv + (size_t)d * C_ * C_, XSb, bv + d * C_, nullptr, Vbf, C_, NS_, C_);

    k_attn<<<dim3(16, 96), 128, attn_smem, stream>>>(
        Qb, Kbf, Vbf, POSb, rpe + (size_t)d * HEADS_ * RPE_W * RPE_W, AOb);

    k_gemm<<<dim3(HW_ / 64, C_ / 64, B_), 256, 0, stream>>>(
        wo + (size_t)d * C_ * C_, AOb, bo + d * C_, X0, X1, C_, HW_, C_);

    k_gemm<<<dim3(HW_ / 64, D2_ / 64, B_), 256, 0, stream>>>(
        w1 + (size_t)d * D2_ * C_, X1, b1 + d * D2_, nullptr, Hb, D2_, HW_, C_);

    int mtot = B_ * D2_ * HW_;
    k_dwconv<<<(mtot + 255) / 256, 256, 0, stream>>>(
        Hb, mdw + (size_t)d * D2_ * 9, mdb + d * D2_, H2b, D2_, 1);

    float* xnext = (d == 1) ? (float*)d_out : X2;
    k_gemm<<<dim3(HW_ / 64, C_ / 64, B_), 256, 0, stream>>>(
        w2 + (size_t)d * C_ * D2_, H2b, b2 + d * C_, X1, xnext, C_, HW_, D2_);
    xprev = xnext;
  }
}
